// Decoder_52106543235146
// MI455X (gfx1250) — matrix operand order repaired and checked by reading, not compiled, not run on hardware
//
#include <hip/hip_runtime.h>
#include <cstddef>

// ---- problem constants (match reference) ----
#define BB    64
#define TT    40
#define NBX   36
#define FSX   1024
#define HSX   512
#define WSX   512
#define VOCX  10000
#define STEPX 20

typedef __attribute__((ext_vector_type(16))) __bf16 v16bf;
typedef __attribute__((ext_vector_type(8)))  __bf16 v8bf;
typedef __attribute__((ext_vector_type(4)))  __bf16 v4bf;
typedef __attribute__((ext_vector_type(2)))  __bf16 v2bf;
typedef __attribute__((ext_vector_type(8)))  float  v8f;
typedef __attribute__((ext_vector_type(4)))  float  v4f;

// ---- WMMA GEMM tiling ----
// 256 threads = 8 waves; block tile 32x64; each wave one 16x16x32 WMMA per K-step.
#define BM 32
#define BN 64
#define BK 32

// Requirements guaranteed by the caller: M % 32 == 0, K % 32 == 0, lda/ldw % 4 == 0,
// 16B-aligned base pointers. Only N may be ragged (NG=true -> guard N in 4-col groups).
template <bool NG>
__device__ __forceinline__ v8f wmma_mainloop(const float* __restrict__ A,
                                             const float* __restrict__ W,
                                             int N, int K, int lda, int ldw,
                                             int blockM, int blockN,
                                             __bf16* As, __bf16* Bs2)
{
  const int tid  = threadIdx.x;
  const int wave = tid >> 5, lane = tid & 31;
  const int wm   = wave >> 2, wn = wave & 3;    // 2x4 wave grid over 32x64 tile
  const int half = lane >> 4, l = lane & 15;

  // staging assignment:
  //   A: thread -> (row = tid>>3, 4 cols at (tid&7)*4)           [1 float4]
  //   B: thread -> (k rows 2*(tid>>4), 2*(tid>>4)+1, 4 cols)     [2 float4]
  const int ar  = tid >> 3, ac4 = (tid & 7) * 4;
  const int bk2 = tid >> 4, bn4 = (tid & 15) * 4;
  const float* aptr = A + (size_t)(blockM + ar) * lda + ac4;
  const float* bptr = W + (size_t)(2 * bk2) * ldw + blockN + bn4;
  const bool bok = !NG || (blockN + bn4 < N);   // N % 4 == 0 -> all-or-none per float4

  v4f ra, rb0 = {0.f,0.f,0.f,0.f}, rb1 = {0.f,0.f,0.f,0.f};
  ra = *(const v4f*)aptr;
  if (bok) { rb0 = *(const v4f*)bptr; rb1 = *(const v4f*)(bptr + ldw); }

  v8f acc = {0.f,0.f,0.f,0.f,0.f,0.f,0.f,0.f};
  for (int k0 = 0; k0 < K; k0 += BK) {
    // ---- store staged regs to LDS (f32 -> bf16 in regs) ----
    {
      v4bf a4;
#pragma unroll
      for (int i = 0; i < 4; ++i) a4[i] = (__bf16)ra[i];
      *(v4bf*)&As[ar * BK + ac4] = a4;                       // ds_store_b64
#pragma unroll
      for (int i = 0; i < 4; ++i) {
        v2bf p; p[0] = (__bf16)rb0[i]; p[1] = (__bf16)rb1[i];
        *(v2bf*)&Bs2[bk2 * 128 + (bn4 + i) * 2] = p;         // ds_store_b32, bank = n%64
      }
    }
    __syncthreads();
    // ---- prefetch next K-tile into regs (overlaps with WMMA below) ----
    if (k0 + BK < K) {
      ra = *(const v4f*)(aptr + k0 + BK);
      if (bok) {
        const float* bp = bptr + (size_t)(k0 + BK) * ldw;
        rb0 = *(const v4f*)bp;
        rb1 = *(const v4f*)(bp + ldw);
      }
    }
    // ---- fragments straight out of LDS in register order ----
    const int am = wm * 16 + l;
    v8bf alo = *(const v8bf*)&As[am * BK + 16 * half];       // ds_load_b128
    v8bf ahi = *(const v8bf*)&As[am * BK + 16 * half + 8];   // ds_load_b128
    v16bf af, bfr;
#pragma unroll
    for (int j = 0; j < 8; ++j) { af[j] = alo[j]; af[8 + j] = ahi[j]; }
    const int bn = wn * 16 + l;
#pragma unroll
    for (int i = 0; i < 8; ++i) {
      v2bf p = *(const v2bf*)&Bs2[(i + 8 * half) * 128 + bn * 2];  // ds_load_b32
      bfr[2 * i] = p[0]; bfr[2 * i + 1] = p[1];
    }
    acc = __builtin_amdgcn_wmma_f32_16x16x32_bf16(false, af, false, bfr,
                                                  (short)0, acc, false, false);
    __syncthreads();
  }
  return acc;
}

// Generic GEMM: C = [C +] A@W + bias, optional tanh. grid = (M/BM, ceil(N/BN))
template <bool NG>
__global__ void gemm_bias_kernel(const float* __restrict__ A, const float* __restrict__ W,
                                 const float* __restrict__ bias, float* __restrict__ C,
                                 int N, int K, int lda, int ldw, int ldc,
                                 int act, int accumulate)
{
  __shared__ __bf16 As[BM * BK];
  __shared__ __bf16 Bs2[BK * BN];
  const int blockM = blockIdx.x * BM;
  const int blockN = blockIdx.y * BN;
  v8f acc = wmma_mainloop<NG>(A, W, N, K, lda, ldw, blockM, blockN, As, Bs2);

  const int tid = threadIdx.x;
  const int wave = tid >> 5, lane = tid & 31;
  const int wm = wave >> 2, wn = wave & 3;
  const int half = lane >> 4, l = lane & 15;
  const int gn = blockN + wn * 16 + l;            // C/D layout: n = lane&15
  if (NG && gn >= N) return;
  const float bv = bias ? bias[gn] : 0.f;
#pragma unroll
  for (int v = 0; v < 8; ++v) {
    int gm = blockM + wm * 16 + v + 8 * half;     // C/D layout: m = v + 8*half
    float r = acc[v] + bv;
    if (accumulate) r += C[(size_t)gm * ldc + gn];
    if (act == 1) r = tanhf(r);
    C[(size_t)gm * ldc + gn] = r;
  }
}

// Fused spatial-attention score GEMM:
// score[row] += sum_n sp_w[n] * tanh(hidq[b][n] + (obj@sp_Wk)[row][n] + sp_bk[n])
__global__ void gemm_spscore_kernel(const float* __restrict__ A, const float* __restrict__ W,
                                    const float* __restrict__ bk, const float* __restrict__ hidq,
                                    const float* __restrict__ spw, float* __restrict__ score,
                                    int N, int K, int lda, int ldw, int TNB)
{
  __shared__ __bf16 As[BM * BK];
  __shared__ __bf16 Bs2[BK * BN];
  __shared__ float  Cs[BM * BN];
  const int blockM = blockIdx.x * BM;
  const int blockN = blockIdx.y * BN;
  v8f acc = wmma_mainloop<false>(A, W, N, K, lda, ldw, blockM, blockN, As, Bs2);

  const int tid = threadIdx.x;
  const int wave = tid >> 5, lane = tid & 31;
  const int wm = wave >> 2, wn = wave & 3;
  const int half = lane >> 4, l = lane & 15;
  const int gn = blockN + wn * 16 + l;
#pragma unroll
  for (int v = 0; v < 8; ++v) {
    int rml = wm * 16 + v + 8 * half;
    int b = (blockM + rml) / TNB;
    float val = tanhf(acc[v] + bk[gn] + hidq[(size_t)b * HSX + gn]) * spw[gn];
    Cs[rml * BN + wn * 16 + l] = val;
  }
  __syncthreads();
  for (int r = tid; r < BM; r += 256) {
    float s = 0.f;
#pragma unroll 8
    for (int c = 0; c < BN; ++c) s += Cs[r * BN + c];
    atomicAdd(&score[blockM + r], s);
  }
}

__global__ void zero_kernel(float* p, int n)
{
  int i = blockIdx.x * blockDim.x + threadIdx.x;
  if (i < n) p[i] = 0.f;
}

// NMS-like phrase redundancy mask. grid = B, block = 128
__global__ void phrmask_kernel(const float* __restrict__ P, int* __restrict__ mask)
{
  __shared__ float Am[STEPX * STEPX];
  __shared__ float Asum[STEPX];
  const int b = blockIdx.x, tid = threadIdx.x;
  for (int i = tid; i < STEPX * STEPX; i += 128) {
    int r = i / STEPX, c = i % STEPX;
    float s = 0.f;
    for (int k = 0; k < STEPX; ++k)
      s += P[((size_t)b * STEPX + r) * STEPX + k] * P[((size_t)b * STEPX + c) * STEPX + k];
    Am[i] = (r == c) ? 0.f : s;
  }
  __syncthreads();
  if (tid < STEPX) {
    float s = 0.f;
    for (int j = 0; j < STEPX; ++j) s += Am[tid * STEPX + j];
    Asum[tid] = s;
  }
  __syncthreads();
  if (tid < STEPX) {
    bool m = false;
    for (int j = tid + 1; j < STEPX; ++j)
      if (Am[tid * STEPX + j] >= 0.2f && !(Asum[j] > Asum[tid])) m = true;
    for (int i = 0; i < tid; ++i)
      if (Am[i * STEPX + tid] >= 0.2f && (Asum[tid] > Asum[i])) m = true;
    mask[b * STEPX + tid] = m ? 1 : 0;
  }
}

// Softmax over boxes + weighted sum of raw box features. grid = B*T, block = 256
__global__ void spattn_kernel(const float* __restrict__ score, const float* __restrict__ obj,
                              float* __restrict__ obj_att)
{
  __shared__ float w[NBX];
  const int bt = blockIdx.x, tid = threadIdx.x;
  if (tid < NBX) w[tid] = score[(size_t)bt * NBX + tid];
  __syncthreads();
  float mx = -1e30f;
  for (int n = 0; n < NBX; ++n) mx = fmaxf(mx, w[n]);
  float sum = 0.f;
  for (int n = 0; n < NBX; ++n) sum += expf(w[n] - mx);
  float inv = 1.f / sum;
  __syncthreads();
  if (tid < NBX) w[tid] = expf(w[tid] - mx) * inv;
  __syncthreads();
  for (int d4 = tid * 4; d4 < FSX; d4 += 1024) {
    v4f acc = {0.f,0.f,0.f,0.f};
    for (int n = 0; n < NBX; ++n) {
      v4f o = *(const v4f*)&obj[((size_t)bt * NBX + n) * FSX + d4];
      float wn = w[n];
#pragma unroll
      for (int u = 0; u < 4; ++u) acc[u] += wn * o[u];
    }
    *(v4f*)&obj_att[(size_t)bt * FSX + d4] = acc;
  }
}

// Additive-attention combine: logits[t] = al_w . tanh(Qp[bs] + Kp[b,t]); softmax_t;
// out = sum a_t * Kraw[b,t].  grid = B*S, block = 256 (wave handles t = wave, wave+8, ...)
__global__ void align_attn_kernel(const float* __restrict__ Qp, int ldq,
                                  const float* __restrict__ Kp,
                                  const float* __restrict__ Kraw, int ldk,
                                  const float* __restrict__ w,
                                  float* __restrict__ out, int ldo,
                                  int S, int Tn, int Dk)
{
  __shared__ float lg[TT];
  const int bs = blockIdx.x;
  const int b  = bs / S;
  const int wave = threadIdx.x >> 5, lane = threadIdx.x & 31;
  const float* qrow = Qp + (size_t)bs * ldq;
  for (int t = wave; t < Tn; t += 8) {
    const float* krow = Kp + ((size_t)b * Tn + t) * HSX;
    float s = 0.f;
    for (int h = lane; h < HSX; h += 32) s += w[h] * tanhf(qrow[h] + krow[h]);
    for (int o = 16; o > 0; o >>= 1) s += __shfl_down(s, o, 32);
    if (lane == 0) lg[t] = s;
  }
  __syncthreads();
  float mx = -1e30f;
  for (int t = 0; t < Tn; ++t) mx = fmaxf(mx, lg[t]);
  float sum = 0.f;
  for (int t = 0; t < Tn; ++t) sum += expf(lg[t] - mx);
  float inv = 1.f / sum;
  __syncthreads();
  if (threadIdx.x < Tn) lg[threadIdx.x] = expf(lg[threadIdx.x] - mx) * inv;
  __syncthreads();
  for (int d4 = threadIdx.x * 4; d4 < Dk; d4 += 1024) {
    v4f acc = {0.f,0.f,0.f,0.f};
    for (int t = 0; t < Tn; ++t) {
      v4f kv = *(const v4f*)&Kraw[((size_t)b * Tn + t) * ldk + d4];
      float a = lg[t];
#pragma unroll
      for (int u = 0; u < 4; ++u) acc[u] += a * kv[u];
    }
    *(v4f*)&out[(size_t)bs * ldo + d4] = acc;
  }
}

// Semantic attention over phrase groups with redundancy mask. grid = B, block = 256.
// Writes feat into decin cols [0,1536) (row stride 2048).
__global__ void sem_attn_kernel(const float* __restrict__ hq, const float* __restrict__ Kg,
                                const float* __restrict__ w, const int* __restrict__ mask,
                                const float* __restrict__ group, float* __restrict__ decin)
{
  __shared__ float lg[STEPX];
  const int b = blockIdx.x;
  const int wave = threadIdx.x >> 5, lane = threadIdx.x & 31;
  const float* qrow = hq + (size_t)b * HSX;
  for (int s = wave; s < STEPX; s += 8) {
    const float* krow = Kg + ((size_t)b * STEPX + s) * HSX;
    float sm = 0.f;
    for (int h = lane; h < HSX; h += 32) sm += w[h] * tanhf(qrow[h] + krow[h]);
    for (int o = 16; o > 0; o >>= 1) sm += __shfl_down(sm, o, 32);
    if (lane == 0) lg[s] = mask[b * STEPX + s] ? -1e9f : sm;
  }
  __syncthreads();
  float mx = -1e30f;
  for (int s = 0; s < STEPX; ++s) mx = fmaxf(mx, lg[s]);
  float sum = 0.f;
  for (int s = 0; s < STEPX; ++s) sum += expf(lg[s] - mx);
  float inv = 1.f / sum;
  __syncthreads();
  if (threadIdx.x < STEPX) lg[threadIdx.x] = expf(lg[threadIdx.x] - mx) * inv;
  __syncthreads();
  for (int d4 = threadIdx.x * 4; d4 < 3 * WSX; d4 += 1024) {
    v4f acc = {0.f,0.f,0.f,0.f};
    for (int s = 0; s < STEPX; ++s) {
      v4f gv = *(const v4f*)&group[((size_t)b * STEPX + s) * (3 * WSX) + d4];
      float a = lg[s];
#pragma unroll
      for (int u = 0; u < 4; ++u) acc[u] += a * gv[u];
    }
    *(v4f*)&decin[(size_t)b * 2048 + d4] = acc;
  }
}

__global__ void copy2d_kernel(float* __restrict__ dst, const float* __restrict__ src,
                              int rows, int cols, int dld, int sld)
{
  int i = blockIdx.x * blockDim.x + threadIdx.x;
  if (i < rows * cols) {
    int r = i / cols, c = i % cols;
    dst[(size_t)r * dld + c] = src[(size_t)r * sld + c];
  }
}

__device__ __forceinline__ float sigm(float x) { return 1.f / (1.f + expf(-x)); }

__global__ void lstm_kernel(const float* __restrict__ gates, const float* __restrict__ c_old,
                            float* __restrict__ hout, float* __restrict__ cout,
                            float* __restrict__ catb)
{
  int i = blockIdx.x * blockDim.x + threadIdx.x;
  if (i < BB * HSX) {
    int b = i / HSX, h = i % HSX;
    const float* g = gates + (size_t)b * 4 * HSX;
    float ig = sigm(g[h]);
    float fg = sigm(g[HSX + h]);
    float gg = tanhf(g[2 * HSX + h]);
    float og = sigm(g[3 * HSX + h]);
    float c  = fg * c_old[i] + ig * gg;
    float hn = og * tanhf(c);
    cout[i] = c;
    hout[i] = hn;
    catb[(size_t)b * 2560 + h] = hn;   // cat = [h_new, dec_in]
  }
}

__global__ void logsoftmax_kernel(const float* __restrict__ x, float* __restrict__ y, int V)
{
  __shared__ float red[256];
  const int b = blockIdx.x, tid = threadIdx.x;
  const float* row = x + (size_t)b * V;
  float mx = -1e30f;
  for (int v = tid; v < V; v += 256) mx = fmaxf(mx, row[v]);
  red[tid] = mx; __syncthreads();
  for (int s = 128; s > 0; s >>= 1) { if (tid < s) red[tid] = fmaxf(red[tid], red[tid + s]); __syncthreads(); }
  mx = red[0]; __syncthreads();
  float sum = 0.f;
  for (int v = tid; v < V; v += 256) sum += expf(row[v] - mx);
  red[tid] = sum; __syncthreads();
  for (int s = 128; s > 0; s >>= 1) { if (tid < s) red[tid] += red[tid + s]; __syncthreads(); }
  float lse = mx + logf(red[0]);
  float* out = y + (size_t)b * V;
  for (int v = tid; v < V; v += 256) out[v] = row[v] - lse;
}

static inline void gemm(hipStream_t s, const float* A, const float* W, const float* bias,
                        float* C, int M, int N, int K, int lda, int ldw, int ldc,
                        int act = 0, int accum = 0)
{
  dim3 g(M / BM, (N + BN - 1) / BN), b(256);
  if (N % BN == 0)
    gemm_bias_kernel<false><<<g, b, 0, s>>>(A, W, bias, C, N, K, lda, ldw, ldc, act, accum);
  else
    gemm_bias_kernel<true><<<g, b, 0, s>>>(A, W, bias, C, N, K, lda, ldw, ldc, act, accum);
}

extern "C" void kernel_launch(void* const* d_in, const int* in_sizes, int n_in,
                              void* d_out, int out_size, void* d_ws, size_t ws_size,
                              hipStream_t stream)
{
  (void)in_sizes; (void)n_in; (void)out_size; (void)ws_size;
  const float* frame = (const float*)d_in[0];
  const float* i3d   = (const float*)d_in[1];
  const float* obj   = (const float*)d_in[2];
  const float* phrA  = (const float*)d_in[3];
  const float* phrF  = (const float*)d_in[4];
  const float* hlast = (const float*)d_in[5];
  const float* cold  = (const float*)d_in[6];
  const float* emb   = (const float*)d_in[7];
  const float* sp_Wq = (const float*)d_in[8];  const float* sp_bq = (const float*)d_in[9];
  const float* sp_Wk = (const float*)d_in[10]; const float* sp_bk = (const float*)d_in[11];
  const float* sp_w  = (const float*)d_in[12];
  const float* al_Wq = (const float*)d_in[13]; const float* al_bq = (const float*)d_in[14];
  const float* al_Wk = (const float*)d_in[15]; const float* al_bk = (const float*)d_in[16];
  const float* al_w  = (const float*)d_in[17];
  const float* sa_Wq = (const float*)d_in[18]; const float* sa_bq = (const float*)d_in[19];
  const float* sa_Wk = (const float*)d_in[20]; const float* sa_bk = (const float*)d_in[21];
  const float* sa_w  = (const float*)d_in[22];
  const float* cnn_W = (const float*)d_in[23]; const float* cnn_b = (const float*)d_in[24];
  const float* mot_W = (const float*)d_in[25]; const float* mot_b = (const float*)d_in[26];
  const float* fra_W = (const float*)d_in[27]; const float* fra_b = (const float*)d_in[28];
  const float* W_ih  = (const float*)d_in[29]; const float* b_ih  = (const float*)d_in[30];
  const float* W_hh  = (const float*)d_in[31]; const float* b_hh  = (const float*)d_in[32];
  const float* out_W = (const float*)d_in[33]; const float* out_b = (const float*)d_in[34];
  const float* wr_W  = (const float*)d_in[35]; const float* wr_b  = (const float*)d_in[36];

  // ---- workspace bump allocator (floats; every size is a multiple of 4 -> 16B alignment) ----
  float* ws = (float*)d_ws;
  size_t off = 0;
  auto alloc = [&](size_t n) { float* p = ws + off; off += n; return p; };
  float* spscore = alloc((size_t)BB * TT * NBX);          // 92160
  int*   mask    = (int*)alloc(BB * STEPX);
  float* hidq    = alloc(BB * HSX);
  float* hq2     = alloc(BB * HSX);
  float* objatt  = alloc((size_t)BB * TT * FSX);
  float* Qp1     = alloc((size_t)BB * STEPX * HSX);
  float* Qp2     = alloc((size_t)BB * STEPX * HSX);
  float* Qp3     = alloc((size_t)BB * STEPX * HSX);
  float* KpO     = alloc((size_t)BB * TT * HSX);
  float* KpI     = alloc((size_t)BB * TT * HSX);
  float* KpF     = alloc((size_t)BB * TT * HSX);
  float* semobj  = alloc((size_t)BB * STEPX * FSX);
  float* semi3d  = alloc((size_t)BB * STEPX * FSX);
  float* semff   = alloc((size_t)BB * STEPX * 2048);      // [sem_fo | sem_fm]
  float* group   = alloc((size_t)BB * STEPX * 1536);      // [phr_frame | phr_mot | phr]
  float* Kg      = alloc((size_t)BB * STEPX * HSX);
  float* decin   = alloc((size_t)BB * 2048);              // [feat | embedded]
  float* gates   = alloc((size_t)BB * 2048);
  float* catb    = alloc((size_t)BB * 2560);              // [h_new | dec_in]
  float* wlog    = alloc((size_t)BB * VOCX);

  float* out_logits = (float*)d_out;
  float* out_h = out_logits + (size_t)BB * VOCX;
  float* out_c = out_h + (size_t)BB * HSX;

  const int MOBJ = BB * TT * NBX;   // 92160
  const int MPH  = BB * STEPX;      // 1280
  const int MBT  = BB * TT;         // 2560

  // 0. score buffer must be zero each call (accumulated via atomics)
  zero_kernel<<<(MOBJ + 255) / 256, 256, 0, stream>>>(spscore, MOBJ);
  // 1. phrase redundancy mask
  phrmask_kernel<<<BB, 128, 0, stream>>>(phrA, mask);
  // 2. hidq = last_hidden @ sp_Wq + sp_bq
  gemm(stream, hlast, sp_Wq, sp_bq, hidq, BB, HSX, HSX, HSX, HSX, HSX);
  // 3. fused: scores = sum_h sp_w*tanh(hidq + obj@sp_Wk + sp_bk)   (the 97-GFLOP GEMM)
  {
    dim3 g(MOBJ / BM, HSX / BN), b(256);
    gemm_spscore_kernel<<<g, b, 0, stream>>>(obj, sp_Wk, sp_bk, hidq, sp_w, spscore,
                                             HSX, FSX, FSX, HSX, TT * NBX);
  }
  // 4. softmax over boxes + combine raw box feats
  spattn_kernel<<<MBT, 256, 0, stream>>>(spscore, obj, objatt);

  // 5. sem_obj = align(phr_feats, obj_att)
  gemm(stream, phrF, al_Wq, al_bq, Qp1, MPH, HSX, WSX, WSX, HSX, HSX);
  gemm(stream, objatt, al_Wk, al_bk, KpO, MBT, HSX, FSX, FSX, HSX, HSX);
  align_attn_kernel<<<MPH, 256, 0, stream>>>(Qp1, HSX, KpO, objatt, FSX, al_w,
                                             semobj, FSX, STEPX, TT, FSX);
  // 6. phr = sem_obj @ cnn_W + cnn_b  -> group cols [1024,1536)
  gemm(stream, semobj, cnn_W, cnn_b, group + 1024, MPH, WSX, FSX, FSX, WSX, 1536);
  // 7. sem_i3d = align(phr, i3d)
  gemm(stream, group + 1024, al_Wq, al_bq, Qp2, MPH, HSX, WSX, 1536, HSX, HSX);
  gemm(stream, i3d, al_Wk, al_bk, KpI, MBT, HSX, FSX, FSX, HSX, HSX);
  align_attn_kernel<<<MPH, 256, 0, stream>>>(Qp2, HSX, KpI, i3d, FSX, al_w,
                                             semi3d, FSX, STEPX, TT, FSX);
  // 8. phr_mot = sem_i3d @ mot_W + mot_b -> group cols [512,1024)
  gemm(stream, semi3d, mot_W, mot_b, group + 512, MPH, WSX, FSX, FSX, WSX, 1536);
  // 9. frame projections + two aligns into semff = [sem_fo | sem_fm]
  gemm(stream, group + 512, al_Wq, al_bq, Qp3, MPH, HSX, WSX, 1536, HSX, HSX);
  gemm(stream, frame, al_Wk, al_bk, KpF, MBT, HSX, FSX, FSX, HSX, HSX);
  align_attn_kernel<<<MPH, 256, 0, stream>>>(Qp2, HSX, KpF, frame, FSX, al_w,
                                             semff, 2048, STEPX, TT, FSX);
  align_attn_kernel<<<MPH, 256, 0, stream>>>(Qp3, HSX, KpF, frame, FSX, al_w,
                                             semff + 1024, 2048, STEPX, TT, FSX);
  // 10. phr_frame = [sem_fo|sem_fm] @ fra_W + fra_b -> group cols [0,512)
  gemm(stream, semff, fra_W, fra_b, group, MPH, WSX, 2048, 2048, WSX, 1536);

  // 11. semantic attention over group with masks -> feat in decin[:, :1536]
  gemm(stream, hlast, sa_Wq, sa_bq, hq2, BB, HSX, HSX, HSX, HSX, HSX);
  gemm(stream, group, sa_Wk, sa_bk, Kg, MPH, HSX, 1536, 1536, HSX, HSX);
  sem_attn_kernel<<<BB, 256, 0, stream>>>(hq2, Kg, sa_w, mask, group, decin);
  // 12. dec_in = [feat, embedded]
  copy2d_kernel<<<(BB * WSX + 255) / 256, 256, 0, stream>>>(decin + 1536, emb, BB, WSX, 2048, WSX);

  // 13. LSTM gates = dec_in@W_ih + b_ih + h@W_hh + b_hh
  gemm(stream, decin, W_ih, b_ih, gates, BB, 2048, 2048, 2048, 2048, 2048, 0, 0);
  gemm(stream, hlast, W_hh, b_hh, gates, BB, 2048, HSX, HSX, 2048, 2048, 0, 1);
  lstm_kernel<<<(BB * HSX + 255) / 256, 256, 0, stream>>>(gates, cold, out_h, out_c, catb);
  // 14. cat = [h_new, dec_in]; dec_out = tanh(cat @ out_W + out_b)  (reuse decin as dec_out)
  copy2d_kernel<<<(BB * 2048 + 255) / 256, 256, 0, stream>>>(catb + 512, decin, BB, 2048, 2560, 2048);
  gemm(stream, catb, out_W, out_b, decin, BB, HSX, 2560, 2560, HSX, HSX, /*tanh*/1, 0);
  // 15. word logits + log_softmax
  gemm(stream, decin, wr_W, wr_b, wlog, BB, VOCX, HSX, HSX, VOCX, VOCX);
  logsoftmax_kernel<<<BB, 256, 0, stream>>>(wlog, out_logits, VOCX);
}
